// Multihead_attention_layer_11811160064309
// MI455X (gfx1250) — matrix operand order repaired and checked by reading, not compiled, not run on hardware
//
#include <hip/hip_runtime.h>
#include <hip/hip_bf16.h>
#include <math.h>

#define EMB 768
#define HEADS 8
#define HEAD_DIM 96
#define BATCH 16
#define SEQ 1024
#define ROWS (BATCH * SEQ) /* 16384 */

#define WLD (EMB + 8)   /* padded LDS row stride for transposed W panel (elems) */
#define SLD (SEQ + 8)   /* padded LDS row stride for fp32 scores (floats)      */
#define PLD (SEQ + 8)   /* padded LDS row stride for bf16 probs (elems)        */

typedef __attribute__((ext_vector_type(16))) __bf16 v16bf;
typedef __attribute__((ext_vector_type(8)))  float  v8f;
typedef __attribute__((ext_vector_type(4)))  float  v4f;
typedef __attribute__((ext_vector_type(4)))  unsigned int v4u;
typedef __attribute__((ext_vector_type(2)))  unsigned int v2u;

static __device__ __forceinline__ unsigned short f32_to_bf16_rte(float f) {
  unsigned int x = __float_as_uint(f);
  unsigned int r = x + 0x7FFFu + ((x >> 16) & 1u);
  return (unsigned short)(r >> 16);
}

union FragU {
  v16bf v;
  v4u   q[2];
};

static __device__ __forceinline__ v8f wmma_bf16(v16bf a, v16bf b, v8f c) {
  return __builtin_amdgcn_wmma_f32_16x16x32_bf16(
      /*neg_a=*/false, a, /*neg_b=*/false, b,
      /*c_mod=*/(short)0, c, /*reuse_a=*/false, /*reuse_b=*/false);
}

// A-matrix 16x32 bf16 fragment from row-major [16][ld] source at column k0.
// CDNA5 layout: lanes 0-15 -> M=lane, K={0..7,16..23}; lanes 16-31 -> M=lane-16,
// K={8..15,24..31}. Each lane's two 8-element K-runs are contiguous -> 2x b128.
static __device__ __forceinline__ v16bf load_frag_a(const unsigned short* base, int ld, int k0) {
  const int lane = threadIdx.x & 31;
  const int row  = lane & 15;
  const int koff = (lane < 16) ? 0 : 8;
  const unsigned short* rp = base + (size_t)row * ld + k0 + koff;
  FragU f;
  f.q[0] = *(const v4u*)(rp);        // K = koff + 0..7   (VGPRs 0-3)
  f.q[1] = *(const v4u*)(rp + 16);   // K = koff + 16..23 (VGPRs 4-7)
  return f.v;
}

// B-matrix 32x16 bf16 fragment where B[kk][n] = src[n*ld + k0 + kk] (N-major source).
// Lanes 0-15 hold K={0..7,16..23}; lanes 16-31 hold K={8..15,24..31} (matches A).
// -> exactly 2x b128 per lane.
static __device__ __forceinline__ v16bf load_frag_bT(const unsigned short* base, int ld, int k0) {
  const int lane = threadIdx.x & 31;
  const int n = lane & 15;
  const unsigned short* rp = base + (size_t)n * ld + k0 + ((lane < 16) ? 0 : 8);
  FragU f;
  f.q[0] = *(const v4u*)(rp);        // K pairs in VGPRs 0-3
  f.q[1] = *(const v4u*)(rp + 16);   // K pairs in VGPRs 4-7
  return f.v;
}

// ---------------- Kernel 1: fp32 -> bf16 conversion of X ----------------
__global__ __launch_bounds__(256) void convert_bf16_kernel(const float* __restrict__ in,
                                                           unsigned short* __restrict__ out,
                                                           int n4) {
  int i = blockIdx.x * 256 + threadIdx.x;
  if (i < n4) {
    v4f f = ((const v4f*)in)[i];
    unsigned int lo = ((unsigned int)f32_to_bf16_rte(f[1]) << 16) | f32_to_bf16_rte(f[0]);
    unsigned int hi = ((unsigned int)f32_to_bf16_rte(f[3]) << 16) | f32_to_bf16_rte(f[2]);
    v2u o; o[0] = lo; o[1] = hi;
    ((v2u*)out)[i] = o;
  }
}

// ---------------- Kernel 2: QKV projection GEMM ----------------
// grid = (EMB/16, ROWS/128, 3), block = 256 (8 waves, each one 16x16 D tile).
// W panel staged TRANSPOSED in LDS so B fragments are 2x ds_load_b128.
// Q scaled by 1/sqrt(Dh). Q,K stored [B,H,S,Dh]; V (z==2) stored row-major into Vrow.
__global__ __launch_bounds__(256) void qkv_proj_kernel(
    const unsigned short* __restrict__ Xb,
    const float* __restrict__ Wq, const float* __restrict__ bq,
    const float* __restrict__ Wk, const float* __restrict__ bk,
    const float* __restrict__ Wv, const float* __restrict__ bv,
    unsigned short* __restrict__ Qb, unsigned short* __restrict__ Kb,
    unsigned short* __restrict__ Vrow) {
  __shared__ unsigned short sWt[16 * WLD];  // transposed panel: row n holds W[:, n0+n]
  const int z = blockIdx.z;
  const float* W    = (z == 0) ? Wq : ((z == 1) ? Wk : Wv);
  const float* bias = (z == 0) ? bq : ((z == 1) ? bk : bv);
  unsigned short* Out = (z == 0) ? Qb : ((z == 1) ? Kb : Vrow);

  const int n0 = blockIdx.x * 16;
  for (int i = threadIdx.x; i < EMB * 16; i += 256) {
    int k = i >> 4, n = i & 15;
    sWt[(size_t)n * WLD + k] = f32_to_bf16_rte(W[(size_t)k * EMB + n0 + n]);
  }
  __syncthreads();

  const int wave = threadIdx.x >> 5;
  const int lane = threadIdx.x & 31;
  const int m0 = blockIdx.y * 128 + wave * 16;

  v8f acc = {};
  for (int k0 = 0; k0 < EMB; k0 += 32) {
    v16bf a = load_frag_a(Xb + (size_t)m0 * EMB, EMB, k0);
    v16bf b = load_frag_bT(sWt, WLD, k0);
    acc = wmma_bf16(a, b, acc);
  }

  const int n  = lane & 15;
  const int mb = (lane < 16) ? 0 : 8;
  const float bval  = bias[n0 + n];
  const float scale = (z == 0) ? 0.10206207261596575f : 1.0f;  // 1/sqrt(96) folded into Q
  const int h  = n0 / HEAD_DIM;
  const int d  = (n0 % HEAD_DIM) + n;
#pragma unroll
  for (int r = 0; r < 8; ++r) {
    int m = m0 + mb + r;
    int bi = m >> 10, s = m & 1023;
    float val = (acc[r] + bval) * scale;
    Out[((size_t)(bi * HEADS + h) * SEQ + s) * HEAD_DIM + d] = f32_to_bf16_rte(val);
  }
}

// ---------------- Kernel 2b: V transpose [bh][s][d] -> Vt [bh][d][s] ----------------
// LDS-tiled so both global read and write are 32-bit+ coalesced. grid = (SEQ/64, B*H).
__global__ __launch_bounds__(256) void transpose_v_kernel(const unsigned short* __restrict__ Vrow,
                                                          unsigned short* __restrict__ Vt) {
  __shared__ unsigned short tile[64 * 100];  // 64 s-rows x 96 d (+4 pad)
  const int s0 = blockIdx.x * 64;
  const int bh = blockIdx.y;
  const unsigned int* src = (const unsigned int*)(Vrow + ((size_t)bh * SEQ + s0) * HEAD_DIM);
  // load 64x96 tile (48 uints per row)
  for (int i = threadIdx.x; i < 64 * 48; i += 256) {
    int r = i / 48, c = i % 48;
    *(unsigned int*)(tile + (size_t)r * 100 + 2 * c) = src[(size_t)r * 48 + c];
  }
  __syncthreads();
  // store transposed: row d holds 64 consecutive s values (32 uints)
  for (int i = threadIdx.x; i < HEAD_DIM * 32; i += 256) {
    int d = i / 32, j = i % 32;
    unsigned int lo = tile[(size_t)(2 * j) * 100 + d];
    unsigned int hi = tile[(size_t)(2 * j + 1) * 100 + d];
    *(unsigned int*)(Vt + ((size_t)bh * HEAD_DIM + d) * SEQ + s0 + 2 * j) = (hi << 16) | lo;
  }
}

// ---------------- Kernel 3: fused scores + softmax + probs-write + context ----------------
// grid = (SEQ/16, B*H), block = 128 (4 waves). Dynamic LDS:
//   float sc[16][SLD] | ushort pb[16][PLD] | float red[16][8] | float rowv[16]
__global__ __launch_bounds__(128) void attention_kernel(
    const unsigned short* __restrict__ Qb, const unsigned short* __restrict__ Kb,
    const unsigned short* __restrict__ Vt, float* __restrict__ probs,
    unsigned short* __restrict__ Ctxb) {
  extern __shared__ char smem[];
  float* sc = (float*)smem;
  unsigned short* pb = (unsigned short*)(sc + 16 * SLD);
  float* red  = (float*)(pb + 16 * PLD);
  float* rowv = red + 16 * 8;

  const int q0 = blockIdx.x * 16;
  const int bh = blockIdx.y;
  const int wave = threadIdx.x >> 5;
  const int lane = threadIdx.x & 31;

  const unsigned short* Qbase  = Qb + ((size_t)bh * SEQ + q0) * HEAD_DIM;
  const unsigned short* Kbase  = Kb + (size_t)bh * SEQ * HEAD_DIM;
  const unsigned short* Vtbase = Vt + (size_t)bh * HEAD_DIM * SEQ;

  // Q fragments reused across all 64 key tiles (Dh = 96 = 3 x 32)
  v16bf qa0 = load_frag_a(Qbase, HEAD_DIM, 0);
  v16bf qa1 = load_frag_a(Qbase, HEAD_DIM, 32);
  v16bf qa2 = load_frag_a(Qbase, HEAD_DIM, 64);

  // ---- scores: each wave owns key tiles kt = wave, wave+4, ... ----
  for (int kt = wave; kt < SEQ / 16; kt += 4) {
    const unsigned short* kb = Kbase + (size_t)kt * 16 * HEAD_DIM;  // N-major for B
    v8f acc = {};
    acc = wmma_bf16(qa0, load_frag_bT(kb, HEAD_DIM, 0), acc);
    acc = wmma_bf16(qa1, load_frag_bT(kb, HEAD_DIM, 32), acc);
    acc = wmma_bf16(qa2, load_frag_bT(kb, HEAD_DIM, 64), acc);
    const int n = lane & 15, mb = (lane < 16) ? 0 : 8;
#pragma unroll
    for (int r = 0; r < 8; ++r) sc[(size_t)(mb + r) * SLD + kt * 16 + n] = acc[r];
  }
  __syncthreads();

  // ---- softmax over 1024 keys: 8 threads per row, 128 elements each (float4) ----
  const int row = threadIdx.x >> 3;
  const int seg = threadIdx.x & 7;
  float* srow = sc + (size_t)row * SLD + seg * 128;
  v4f* s4 = (v4f*)srow;

  float mx = -INFINITY;
  for (int i = 0; i < 32; ++i) {
    v4f t = s4[i];
    mx = fmaxf(mx, fmaxf(fmaxf(t[0], t[1]), fmaxf(t[2], t[3])));
  }
  red[row * 8 + seg] = mx;
  __syncthreads();
  if (seg == 0) {
    float m = red[row * 8];
    for (int i = 1; i < 8; ++i) m = fmaxf(m, red[row * 8 + i]);
    rowv[row] = m;
  }
  __syncthreads();
  const float rmax = rowv[row];
  float sum = 0.f;
  for (int i = 0; i < 32; ++i) {
    v4f t = s4[i];
    t[0] = __expf(t[0] - rmax); t[1] = __expf(t[1] - rmax);
    t[2] = __expf(t[2] - rmax); t[3] = __expf(t[3] - rmax);
    s4[i] = t;
    sum += (t[0] + t[1]) + (t[2] + t[3]);
  }
  __syncthreads();
  red[row * 8 + seg] = sum;
  __syncthreads();
  if (seg == 0) {
    float s = 0.f;
    for (int i = 0; i < 8; ++i) s += red[row * 8 + i];
    rowv[row] = s;
  }
  __syncthreads();
  const float inv = 1.0f / rowv[row];
  v4f* pout4 = (v4f*)(probs + ((size_t)bh * SEQ + q0 + row) * SEQ + seg * 128);
  v2u* pbr2  = (v2u*)(pb + (size_t)row * PLD + seg * 128);
  for (int i = 0; i < 32; ++i) {
    v4f t = s4[i];
    t[0] *= inv; t[1] *= inv; t[2] *= inv; t[3] *= inv;
    pout4[i] = t;  // fp32 probs output: 537 MB streamed as b128 stores
    v2u o;
    o[0] = ((unsigned int)f32_to_bf16_rte(t[1]) << 16) | f32_to_bf16_rte(t[0]);
    o[1] = ((unsigned int)f32_to_bf16_rte(t[3]) << 16) | f32_to_bf16_rte(t[2]);
    pbr2[i] = o;  // bf16 copy for context WMMA
  }
  __syncthreads();

  // ---- context = P[16,1024] x V[1024,96]: 6 N-tiles split over 4 waves ----
  const int bi = bh >> 3, h = bh & 7;
  for (int nt = wave; nt < HEAD_DIM / 16; nt += 4) {
    const unsigned short* vb = Vtbase + (size_t)nt * 16 * SEQ;  // N-major (transposed V)
    v8f acc = {};
    for (int k0 = 0; k0 < SEQ; k0 += 32) {
      v16bf a = load_frag_a(pb, PLD, k0);
      v16bf b = load_frag_bT(vb, SEQ, k0);
      acc = wmma_bf16(a, b, acc);
    }
    const int n = lane & 15, mb = (lane < 16) ? 0 : 8;
#pragma unroll
    for (int r = 0; r < 8; ++r) {
      Ctxb[((size_t)bi * SEQ + q0 + mb + r) * EMB + h * HEAD_DIM + nt * 16 + n] =
          f32_to_bf16_rte(acc[r]);
    }
  }
}

// ---------------- Kernel 4: output projection GEMM (fp32 out + bias) ----------------
__global__ __launch_bounds__(256) void out_proj_kernel(const unsigned short* __restrict__ Ctxb,
                                                       const float* __restrict__ Wo,
                                                       const float* __restrict__ bo,
                                                       float* __restrict__ Out) {
  __shared__ unsigned short sWt[16 * WLD];
  const int n0 = blockIdx.x * 16;
  for (int i = threadIdx.x; i < EMB * 16; i += 256) {
    int k = i >> 4, n = i & 15;
    sWt[(size_t)n * WLD + k] = f32_to_bf16_rte(Wo[(size_t)k * EMB + n0 + n]);
  }
  __syncthreads();

  const int wave = threadIdx.x >> 5;
  const int lane = threadIdx.x & 31;
  const int m0 = blockIdx.y * 128 + wave * 16;

  v8f acc = {};
  for (int k0 = 0; k0 < EMB; k0 += 32) {
    v16bf a = load_frag_a(Ctxb + (size_t)m0 * EMB, EMB, k0);
    v16bf b = load_frag_bT(sWt, WLD, k0);
    acc = wmma_bf16(a, b, acc);
  }
  const int n = lane & 15, mb = (lane < 16) ? 0 : 8;
  const float bval = bo[n0 + n];
#pragma unroll
  for (int r = 0; r < 8; ++r) {
    Out[(size_t)(m0 + mb + r) * EMB + n0 + n] = acc[r] + bval;
  }
}

extern "C" void kernel_launch(void* const* d_in, const int* in_sizes, int n_in,
                              void* d_out, int out_size, void* d_ws, size_t ws_size,
                              hipStream_t stream) {
  const float* X  = (const float*)d_in[0];
  const float* Wq = (const float*)d_in[1];
  const float* bq = (const float*)d_in[2];
  const float* Wk = (const float*)d_in[3];
  const float* bk = (const float*)d_in[4];
  const float* Wv = (const float*)d_in[5];
  const float* bv = (const float*)d_in[6];
  const float* Wo = (const float*)d_in[7];
  const float* bo = (const float*)d_in[8];

  float* out   = (float*)d_out;
  float* probs = out + (size_t)BATCH * SEQ * EMB;  // second output in tuple order

  // Workspace: 5 bf16 buffers of ROWS*EMB elements (~126 MB total).
  // Cb doubles as row-major V staging before the transpose (attention overwrites it after).
  const size_t elems = (size_t)ROWS * EMB;
  unsigned short* Xb = (unsigned short*)d_ws;
  unsigned short* Qb = Xb + elems;
  unsigned short* Kb = Qb + elems;
  unsigned short* Vb = Kb + elems;  // holds Vt after transpose
  unsigned short* Cb = Vb + elems;  // V row-major staging, then context

  const int n4 = (int)(elems / 4);
  hipLaunchKernelGGL(convert_bf16_kernel, dim3((n4 + 255) / 256), dim3(256), 0, stream,
                     X, Xb, n4);

  hipLaunchKernelGGL(qkv_proj_kernel, dim3(EMB / 16, ROWS / 128, 3), dim3(256), 0, stream,
                     Xb, Wq, bq, Wk, bk, Wv, bv, Qb, Kb, /*Vrow=*/Cb);

  hipLaunchKernelGGL(transpose_v_kernel, dim3(SEQ / 64, BATCH * HEADS), dim3(256), 0, stream,
                     /*Vrow=*/Cb, /*Vt=*/Vb);

  const size_t attn_lds = (size_t)16 * SLD * sizeof(float)
                        + (size_t)16 * PLD * sizeof(unsigned short)
                        + 16 * 8 * sizeof(float) + 16 * sizeof(float);
  hipLaunchKernelGGL(attention_kernel, dim3(SEQ / 16, BATCH * HEADS), dim3(128), attn_lds,
                     stream, Qb, Kb, Vb, probs, Cb);

  hipLaunchKernelGGL(out_proj_kernel, dim3(EMB / 16, ROWS / 128), dim3(256), 0, stream,
                     Cb, Wo, bo, out);
}
